// GNNPolicy_50663434224371
// MI455X (gfx1250) — matrix operand order repaired and checked by reading, not compiled, not run on hardware
//
#include <hip/hip_runtime.h>
#include <hip/hip_bf16.h>

// ---------------------------------------------------------------------------
// GNN forward for MI455X (gfx1250, wave32, WMMA).
// - All 64/128-deep linears via v_wmma_f32_16x16x32_f16 (f32 accumulate).
// - Node activations stored f16 (halves L2 gather traffic; edge gathers are
//   the bottleneck: 1M edges x 2 x 128B, L2-resident).
// - A-tile staging uses CDNA5 async global->LDS copies (ASYNCcnt) since with
//   f16 activations it is a pure byte copy; f32 only for LN stats + agg atomics.
// ---------------------------------------------------------------------------

typedef __attribute__((ext_vector_type(16))) _Float16 v16h;
typedef __attribute__((ext_vector_type(8)))  float    v8f;

// A-fragment (16x32 f16, MxK): lane holds row (lane&15); VGPR v holds K pair
//   K = 16*hi + 2*v   (contiguous, matching frag_b)
static __device__ __forceinline__ v16h frag_a(const _Float16* rowbase, int hi) {
  const _Float16* p = rowbase + 16 * hi;
  v16h a;
#pragma unroll
  for (int v = 0; v < 8; ++v) {
    int kk = 2 * v;
    a[2 * v]     = p[kk];
    a[2 * v + 1] = p[kk + 1];
  }
  return a;
}

// B-fragment (32x16 f16, KxN): lane holds col; lanes 0-15 K=0..15,
// lanes 16-31 K=16..31 (contiguous, same k order as frag_a); 16 contiguous halves.
static __device__ __forceinline__ v16h frag_b(const _Float16* rowbase, int hi) {
  const _Float16* p = rowbase + 16 * hi;
  v16h b;
#pragma unroll
  for (int i = 0; i < 16; ++i) b[i] = p[i];
  return b;
}

static __device__ __forceinline__ v8f wmma_f16(v16h a, v16h b, v8f c) {
  return __builtin_amdgcn_wmma_f32_16x16x32_f16(false, a, false, b, (short)0, c,
                                                false, false);
}

// CDNA5 async global->LDS 16B copy (VDST = LDS byte offset = low 32 bits of
// the generic shared pointer; VADDR = 64-bit global address; SADDR = off).
static __device__ __forceinline__ void async_g2l_b128(void* lds, const void* g) {
  unsigned loff = (unsigned)(unsigned long long)lds;
  asm volatile("global_load_async_to_lds_b128 %0, %1, off"
               :: "v"(loff), "v"(g) : "memory");
}
static __device__ __forceinline__ void wait_async0() {
  asm volatile("s_wait_asynccnt 0x0" ::: "memory");
}

// ---------------------------------------------------------------------------
// Y[N,64] = act(X[N,64] @ W[64,64]^T + bias)   (X,Y,W all f16; f32 accum)
// 256 threads, 64 rows per block, 8 waves -> 4 row-tiles x 4 col-tiles.
// ---------------------------------------------------------------------------
template <bool RELU>
__global__ void k_linear64(const _Float16* __restrict__ X,
                           const _Float16* __restrict__ W,
                           const float* __restrict__ bias,
                           _Float16* __restrict__ Y, int N) {
  __shared__ _Float16 sA[64 * 64];
  int t = threadIdx.x;
  int row0 = blockIdx.x * 64;
  bool full = (row0 + 64 <= N);
  if (full) {
    // pure byte copy: 8KB staged via async global->LDS, 32B per thread
    const char* src = (const char*)(X + (size_t)row0 * 64);
    char* dst = (char*)sA;
    async_g2l_b128(dst + t * 32,      src + t * 32);
    async_g2l_b128(dst + t * 32 + 16, src + t * 32 + 16);
    wait_async0();
  } else {
    int rl = t >> 2, c0 = (t & 3) * 16;
    int gr = row0 + rl;
    bool ok = gr < N;
    const _Float16* xp = X + (size_t)(ok ? gr : N - 1) * 64 + c0;
#pragma unroll
    for (int j = 0; j < 16; ++j)
      sA[rl * 64 + c0 + j] = ok ? xp[j] : (_Float16)0.f;
  }
  __syncthreads();
  int wave = t >> 5, lane = t & 31, hi = lane >> 4, l15 = lane & 15;
  int rt = wave >> 1, cb = (wave & 1) * 32;
#pragma unroll
  for (int ct = 0; ct < 2; ++ct) {
    int col0 = cb + ct * 16;
    v8f acc = {};
#pragma unroll
    for (int kc = 0; kc < 2; ++kc) {
      v16h a = frag_a(&sA[(rt * 16 + l15) * 64 + kc * 32], hi);
      v16h b = frag_b(&W[(col0 + l15) * 64 + kc * 32], hi);
      acc = wmma_f16(a, b, acc);
    }
    float bv = bias[col0 + l15];
    int r0 = row0 + rt * 16 + 8 * hi;
    if (full) {
      _Float16* yp = Y + (size_t)r0 * 64 + col0 + l15;
#pragma unroll
      for (int v = 0; v < 8; ++v) {
        float r = acc[v] + bv;
        if (RELU) r = fmaxf(r, 0.f);
        yp[v * 64] = (_Float16)r;
      }
    } else {
#pragma unroll
      for (int v = 0; v < 8; ++v) {
        int gr = r0 + v;
        if (gr < N) {
          float r = acc[v] + bv;
          if (RELU) r = fmaxf(r, 0.f);
          Y[(size_t)gr * 64 + col0 + l15] = (_Float16)r;
        }
      }
    }
  }
}

// ---------------------------------------------------------------------------
// Fused edge kernel: m = tl[tgt] + sr[src] + ew ; LN(g1,be1) ; relu ;
// m @ Wf^T + bf ; atomic scatter-add into agg[tgt].  64 edges per block.
// ---------------------------------------------------------------------------
__global__ void k_edge(const _Float16* __restrict__ tl,
                       const _Float16* __restrict__ sr,
                       const float* __restrict__ ew, const float* __restrict__ g1,
                       const float* __restrict__ be1,
                       const _Float16* __restrict__ Wf,
                       const float* __restrict__ bf,
                       const int* __restrict__ srcIdx,
                       const int* __restrict__ tgtIdx, int E,
                       float* __restrict__ agg) {
  __shared__ _Float16 sM[64 * 64];
  __shared__ int sTgt[64];
  int t = threadIdx.x;
  int e0 = blockIdx.x * 64;
  bool full = (e0 + 64 <= E);
  {
    int ei = t >> 2, part = t & 3;
    int eg = e0 + ei;
    bool ok = full || (eg < E);
    float m[16];
    if (ok) {
      int tg = tgtIdx[eg], sn = srcIdx[eg];
      if (part == 0) sTgt[ei] = tg;
      const _Float16* tp = tl + (size_t)tg * 64 + part * 16;  // 32B gather
      const _Float16* sp = sr + (size_t)sn * 64 + part * 16;  // 32B gather
#pragma unroll
      for (int j = 0; j < 16; ++j)
        m[j] = (float)tp[j] + (float)sp[j] + ew[part * 16 + j];
    } else {
      if (part == 0) sTgt[ei] = 0;
#pragma unroll
      for (int j = 0; j < 16; ++j) m[j] = 0.f;
    }
    float s = 0.f, ss = 0.f;
#pragma unroll
    for (int j = 0; j < 16; ++j) { s += m[j]; ss += m[j] * m[j]; }
    s  += __shfl_xor(s, 1);  s  += __shfl_xor(s, 2);   // 4 lanes share an edge
    ss += __shfl_xor(ss, 1); ss += __shfl_xor(ss, 2);
    float mean = s * (1.f / 64.f);
    float var  = ss * (1.f / 64.f) - mean * mean;
    float rstd = rsqrtf(var + 1e-5f);
#pragma unroll
    for (int j = 0; j < 16; ++j) {
      float h = (m[j] - mean) * rstd * g1[part * 16 + j] + be1[part * 16 + j];
      sM[ei * 64 + part * 16 + j] = (_Float16)fmaxf(h, 0.f);
    }
  }
  __syncthreads();
  int wave = t >> 5, lane = t & 31, hi = lane >> 4, l15 = lane & 15;
  int rt = wave >> 1, cb = (wave & 1) * 32;
  // hoist target rows for this lane's 8 edges out of the column loop
  unsigned tgv[8];
#pragma unroll
  for (int v = 0; v < 8; ++v) tgv[v] = (unsigned)sTgt[rt * 16 + 8 * hi + v];
#pragma unroll
  for (int ct = 0; ct < 2; ++ct) {
    int col0 = cb + ct * 16;
    v8f acc = {};
#pragma unroll
    for (int kc = 0; kc < 2; ++kc) {
      v16h a = frag_a(&sM[(rt * 16 + l15) * 64 + kc * 32], hi);
      v16h b = frag_b(&Wf[(col0 + l15) * 64 + kc * 32], hi);
      acc = wmma_f16(a, b, acc);
    }
    float bv = bf[col0 + l15];
    if (full) {
#pragma unroll
      for (int v = 0; v < 8; ++v)
        atomicAdd(agg + (tgv[v] * 64u + (unsigned)(col0 + l15)), acc[v] + bv);
    } else {
      int er0 = e0 + rt * 16 + 8 * hi;
#pragma unroll
      for (int v = 0; v < 8; ++v) {
        if (er0 + v < E)
          atomicAdd(agg + (tgv[v] * 64u + (unsigned)(col0 + l15)), acc[v] + bv);
      }
    }
  }
}

// ---------------------------------------------------------------------------
// Y = relu( concat[ LN(agg,g2,be2), right ] @ Wo1[64,128]^T + bo1 )
// ---------------------------------------------------------------------------
__global__ void k_concat_lin(const float* __restrict__ agg,
                             const float* __restrict__ g2,
                             const float* __restrict__ be2,
                             const _Float16* __restrict__ right,
                             const _Float16* __restrict__ Wo1,
                             const float* __restrict__ bo1,
                             _Float16* __restrict__ Y, int N) {
  __shared__ _Float16 sH[64 * 128];
  int t = threadIdx.x;
  int row0 = blockIdx.x * 64;
  bool full = (row0 + 64 <= N);
  {
    int rl = t >> 2, part = t & 3;
    int gr = row0 + rl;
    bool ok = full || (gr < N);
    int gc = ok ? gr : N - 1;
    float a[16];
    const float* ap = agg + (size_t)gc * 64 + part * 16;
#pragma unroll
    for (int j = 0; j < 16; ++j) a[j] = ok ? ap[j] : 0.f;
    float s = 0.f, ss = 0.f;
#pragma unroll
    for (int j = 0; j < 16; ++j) { s += a[j]; ss += a[j] * a[j]; }
    s  += __shfl_xor(s, 1);  s  += __shfl_xor(s, 2);
    ss += __shfl_xor(ss, 1); ss += __shfl_xor(ss, 2);
    float mean = s * (1.f / 64.f);
    float var  = ss * (1.f / 64.f) - mean * mean;
    float rstd = rsqrtf(var + 1e-5f);
    const _Float16* rp = right + (size_t)gc * 64 + part * 16;
#pragma unroll
    for (int j = 0; j < 16; ++j) {
      sH[rl * 128 + part * 16 + j] =
          (_Float16)((a[j] - mean) * rstd * g2[part * 16 + j] + be2[part * 16 + j]);
      sH[rl * 128 + 64 + part * 16 + j] = ok ? rp[j] : (_Float16)0.f;
    }
  }
  __syncthreads();
  int wave = t >> 5, lane = t & 31, hi = lane >> 4, l15 = lane & 15;
  int rt = wave >> 1, cb = (wave & 1) * 32;
#pragma unroll
  for (int ct = 0; ct < 2; ++ct) {
    int col0 = cb + ct * 16;
    v8f acc = {};
#pragma unroll
    for (int kc = 0; kc < 4; ++kc) {
      v16h a = frag_a(&sH[(rt * 16 + l15) * 128 + kc * 32], hi);
      v16h b = frag_b(&Wo1[(col0 + l15) * 128 + kc * 32], hi);
      acc = wmma_f16(a, b, acc);
    }
    float bv = bo1[col0 + l15];
    int r0 = row0 + rt * 16 + 8 * hi;
    if (full) {
      _Float16* yp = Y + (size_t)r0 * 64 + col0 + l15;
#pragma unroll
      for (int v = 0; v < 8; ++v) yp[v * 64] = (_Float16)fmaxf(acc[v] + bv, 0.f);
    } else {
#pragma unroll
      for (int v = 0; v < 8; ++v) {
        int gr = r0 + v;
        if (gr < N)
          Y[(size_t)gr * 64 + col0 + l15] = (_Float16)fmaxf(acc[v] + bv, 0.f);
      }
    }
  }
}

// ---------------------------------------------------------------------------
// Y[N,16] (f32) = X[N,64] @ W3[16,64]^T + b3   (final heads; 128 rows/block)
// ---------------------------------------------------------------------------
__global__ void k_out16(const _Float16* __restrict__ X,
                        const _Float16* __restrict__ W3,
                        const float* __restrict__ b3,
                        float* __restrict__ Y, int N) {
  __shared__ _Float16 sA[128 * 64];
  int t = threadIdx.x;
  int row0 = blockIdx.x * 128;
  bool full = (row0 + 128 <= N);
  if (full) {
    // 16KB staged via async global->LDS, 64B per thread
    const char* src = (const char*)(X + (size_t)row0 * 64);
    char* dst = (char*)sA;
#pragma unroll
    for (int q = 0; q < 4; ++q)
      async_g2l_b128(dst + t * 64 + q * 16, src + t * 64 + q * 16);
    wait_async0();
  } else {
    int rl = t >> 1, c0 = (t & 1) * 32;
    int gr = row0 + rl;
    bool ok = gr < N;
    const _Float16* xp = X + (size_t)(ok ? gr : N - 1) * 64 + c0;
#pragma unroll
    for (int j = 0; j < 32; ++j)
      sA[rl * 64 + c0 + j] = ok ? xp[j] : (_Float16)0.f;
  }
  __syncthreads();
  int wave = t >> 5, lane = t & 31, hi = lane >> 4, l15 = lane & 15;
  int rt = wave;
  v8f acc = {};
#pragma unroll
  for (int kc = 0; kc < 2; ++kc) {
    v16h a = frag_a(&sA[(rt * 16 + l15) * 64 + kc * 32], hi);
    v16h b = frag_b(&W3[l15 * 64 + kc * 32], hi);
    acc = wmma_f16(a, b, acc);
  }
  float bv = b3[l15];
  int r0 = row0 + rt * 16 + 8 * hi;
  if (full) {
    float* yp = Y + (size_t)r0 * 16 + l15;
#pragma unroll
    for (int v = 0; v < 8; ++v) yp[v * 16] = acc[v] + bv;
  } else {
#pragma unroll
    for (int v = 0; v < 8; ++v) {
      int gr = r0 + v;
      if (gr < N) Y[(size_t)gr * 16 + l15] = acc[v] + bv;
    }
  }
}

// ---------------------------------------------------------------------------
// Embedding first stage: LN(in_d) + Linear(in_d->64) + relu (in_d = 3 or 5)
// ---------------------------------------------------------------------------
__global__ void k_embed_first(const float* __restrict__ X,
                              const float* __restrict__ g,
                              const float* __restrict__ b,
                              const float* __restrict__ W1,
                              const float* __restrict__ b1,
                              _Float16* __restrict__ Y, int N, int in_d) {
  int n = blockIdx.x * 256 + threadIdx.x;
  if (n >= N) return;
  float x[8], h[8];
  float s = 0.f;
  for (int d = 0; d < in_d; ++d) { x[d] = X[(size_t)n * in_d + d]; s += x[d]; }
  float mean = s / (float)in_d;
  float vs = 0.f;
  for (int d = 0; d < in_d; ++d) vs += (x[d] - mean) * (x[d] - mean);
  float rstd = rsqrtf(vs / (float)in_d + 1e-5f);
  for (int d = 0; d < in_d; ++d) h[d] = (x[d] - mean) * rstd * g[d] + b[d];
  for (int o = 0; o < 64; ++o) {
    float acc = b1[o];
    for (int d = 0; d < in_d; ++d) acc += h[d] * W1[o * in_d + d];
    Y[(size_t)n * 64 + o] = (_Float16)fmaxf(acc, 0.f);
  }
}

__global__ void k_cvt_f16(const float* __restrict__ s, _Float16* __restrict__ d, int n) {
  int i = blockIdx.x * 256 + threadIdx.x;
  if (i < n) d[i] = (_Float16)s[i];
}
__global__ void k_zero(float* __restrict__ p, long n) {
  long i = (long)blockIdx.x * 1024 + threadIdx.x;
  if (i < n) p[i] = 0.f;
}
// LN over a single edge feature is exactly edge_b; fold into ew = edge_b*We[:,0]
__global__ void k_ew(const float* __restrict__ We, const float* __restrict__ edge_b,
                     float* __restrict__ ew) {
  int f = threadIdx.x;
  if (f < 64) ew[f] = edge_b[0] * We[f];
}

// ---------------------------------------------------------------------------
extern "C" void kernel_launch(void* const* d_in, const int* in_sizes, int n_in,
                              void* d_out, int out_size, void* d_ws, size_t ws_size,
                              hipStream_t stream) {
  (void)n_in; (void)out_size; (void)ws_size;
  const int NC = in_sizes[0] / 3;
  const int NV = in_sizes[3] / 5;
  const int E  = in_sizes[1] / 2;
  const float* cons_x = (const float*)d_in[0];
  const int*   c_idx  = (const int*)d_in[1];       // edge_indices row 0 (constraints)
  const int*   v_idx  = c_idx + E;                 // edge_indices row 1 (variables)
  const float* var_x  = (const float*)d_in[3];
  auto P = [&](int i) { return (const float*)d_in[i]; };

  // Param leaf indices: jax pytree flatten, dict keys sorted alphabetically.
  // 4..9   con_out  {W1,W2,W3,b1,b2,b3}
  // 10..15 cons_emb {W1,W2,b,b1,b2,g}
  // 16..29 conv_cv  {We,Wf,Wl,Wo1,Wo2,Wr,be1,be2,bf,bl,bo1,bo2,g1,g2}
  // 30..43 conv_vc  (same order)
  // 44 edge_b, 45 edge_g
  // 46..51 var_emb {W1,W2,b,b1,b2,g}
  // 52..57 var_out {W1,W2,W3,b1,b2,b3}

  char* base = (char*)d_ws;
  size_t off = 0;
  auto alloc = [&](size_t bytes) -> void* {
    void* p = base + off;
    off = (off + bytes + 255) & ~(size_t)255;
    return p;
  };
  auto cvt = [&](const float* src, int n) -> _Float16* {
    _Float16* d = (_Float16*)alloc((size_t)n * sizeof(_Float16));
    k_cvt_f16<<<(n + 255) / 256, 256, 0, stream>>>(src, d, n);
    return d;
  };

  // f16 weight caches
  _Float16* w2c = cvt(P(11), 4096);            // cons_emb.W2
  _Float16* w2v = cvt(P(47), 4096);            // var_emb.W2
  struct Conv {
    _Float16 *wl, *wr, *wf, *wo1, *wo2;
    const float *bl, *bf, *bo1, *bo2, *g1, *be1, *g2, *be2;
    float* ew;
  };
  auto make_conv = [&](int b0) -> Conv {
    Conv c;
    c.wf  = cvt(P(b0 + 1), 4096);              // Wf
    c.wl  = cvt(P(b0 + 2), 4096);              // Wl
    c.wo1 = cvt(P(b0 + 3), 8192);              // Wo1 [64,128]
    c.wo2 = cvt(P(b0 + 4), 4096);              // Wo2
    c.wr  = cvt(P(b0 + 5), 4096);              // Wr
    c.be1 = P(b0 + 6);  c.be2 = P(b0 + 7);
    c.bf  = P(b0 + 8);  c.bl  = P(b0 + 9);
    c.bo1 = P(b0 + 10); c.bo2 = P(b0 + 11);
    c.g1  = P(b0 + 12); c.g2  = P(b0 + 13);
    c.ew  = (float*)alloc(64 * sizeof(float));
    k_ew<<<1, 64, 0, stream>>>(P(b0 + 0), P(44), c.ew);  // We, edge_b
    return c;
  };
  Conv cv = make_conv(16);  // conv_cv  (c -> v)
  Conv vc = make_conv(30);  // conv_vc  (v -> c)
  _Float16* co_w1 = cvt(P(4), 4096);
  _Float16* co_w2 = cvt(P(5), 4096);
  _Float16* co_w3 = cvt(P(6), 1024);
  _Float16* vo_w1 = cvt(P(52), 4096);
  _Float16* vo_w2 = cvt(P(53), 4096);
  _Float16* vo_w3 = cvt(P(54), 1024);

  // zero bias vector (for the bias-free Wr linear)
  float* zb = (float*)alloc(64 * sizeof(float));
  k_zero<<<1, 64, 0, stream>>>(zb, 64);

  // node buffers: activations f16 (L2-resident), agg f32 (atomic target)
  _Float16* cA  = (_Float16*)alloc((size_t)NC * 64 * 2);
  _Float16* cB  = (_Float16*)alloc((size_t)NC * 64 * 2);
  _Float16* vA  = (_Float16*)alloc((size_t)NV * 64 * 2);
  _Float16* vB  = (_Float16*)alloc((size_t)NV * 64 * 2);
  _Float16* tl  = (_Float16*)alloc((size_t)NV * 64 * 2);  // reused as h1 / tmp
  _Float16* sr  = (_Float16*)alloc((size_t)NV * 64 * 2);
  float*    agg = (float*)alloc((size_t)NV * 64 * 4);

  // --- embeddings ---
  k_embed_first<<<(NC + 255) / 256, 256, 0, stream>>>(cons_x, P(15), P(12), P(10), P(13), tl, NC, 3);
  k_linear64<true><<<(NC + 63) / 64, 256, 0, stream>>>(tl, w2c, P(14), cA, NC);
  k_embed_first<<<(NV + 255) / 256, 256, 0, stream>>>(var_x, P(51), P(48), P(46), P(49), tl, NV, 5);
  k_linear64<true><<<(NV + 63) / 64, 256, 0, stream>>>(tl, w2v, P(50), vA, NV);

  auto biconv = [&](const Conv& p, const _Float16* left, int leftN, const int* srcI,
                    const int* tgtI, const _Float16* right, int rightN, _Float16* out) {
    k_linear64<false><<<(rightN + 63) / 64, 256, 0, stream>>>(right, p.wl, p.bl, tl, rightN);
    k_linear64<false><<<(leftN + 63) / 64, 256, 0, stream>>>(left, p.wr, zb, sr, leftN);
    long zn = (long)rightN * 64;
    k_zero<<<(int)((zn + 1023) / 1024), 1024, 0, stream>>>(agg, zn);
    k_edge<<<(E + 63) / 64, 256, 0, stream>>>(tl, sr, p.ew, p.g1, p.be1,
                                              p.wf, p.bf, srcI, tgtI, E, agg);
    k_concat_lin<<<(rightN + 63) / 64, 256, 0, stream>>>(agg, p.g2, p.be2, right,
                                                         p.wo1, p.bo1, tl, rightN);
    k_linear64<false><<<(rightN + 63) / 64, 256, 0, stream>>>(tl, p.wo2, p.bo2, out, rightN);
  };

  _Float16 *c = cA, *cAlt = cB, *v = vA, *vAlt = vB;
  for (int it = 0; it < 3; ++it) {
    biconv(vc, v, NV, v_idx, c_idx, c, NC, cAlt);
    { _Float16* tmp = c; c = cAlt; cAlt = tmp; }
    biconv(cv, c, NC, c_idx, v_idx, v, NV, vAlt);
    { _Float16* tmp = v; v = vAlt; vAlt = tmp; }
  }

  // --- output heads ---
  float* out = (float*)d_out;
  k_linear64<true><<<(NC + 63) / 64, 256, 0, stream>>>(c, co_w1, P(7), tl, NC);
  k_linear64<true><<<(NC + 63) / 64, 256, 0, stream>>>(tl, co_w2, P(8), sr, NC);
  k_out16<<<(NC + 127) / 128, 256, 0, stream>>>(sr, co_w3, P(9), out, NC);

  k_linear64<true><<<(NV + 63) / 64, 256, 0, stream>>>(v, vo_w1, P(55), tl, NV);
  k_linear64<true><<<(NV + 63) / 64, 256, 0, stream>>>(tl, vo_w2, P(56), sr, NV);
  k_out16<<<(NV + 127) / 128, 256, 0, stream>>>(sr, vo_w3, P(57), out + (size_t)NC * 16, NV);
}
